// DAGCNLayer_16947940950535
// MI455X (gfx1250) — matrix operand order repaired and checked by reading, not compiled, not run on hardware
//
#include <hip/hip_runtime.h>
#include <math.h>

// ---------------------------------------------------------------------------
// CDNA5 (gfx1250) wave32 WMMA implementation of the DAG-transformer layer.
// All dense math runs through v_wmma_f32_16x16x32_f16 (f16 in, f32 accum).
// LDS tile layouts are chosen so every WMMA fragment load is a contiguous
// 16-byte run per lane (ds_load_b128), not per-element ds_load_u16 gathers.
// ---------------------------------------------------------------------------

typedef __attribute__((ext_vector_type(16))) _Float16 v16h;
typedef __attribute__((ext_vector_type(8)))  _Float16 v8h;
typedef __attribute__((ext_vector_type(4)))  _Float16 v4h;
typedef __attribute__((ext_vector_type(8)))  float    v8f;
typedef __attribute__((ext_vector_type(4)))  float    v4f;

#define WMMA16(a, b, c) \
  __builtin_amdgcn_wmma_f32_16x16x32_f16(false, (a), false, (b), (short)0, (c), false, false)

#define N_NODES 4096
#define H_DIM   256
#define N_HEADS 4
#define DH      64
#define E_EDGES 262144

__device__ __forceinline__ float gelu_exact(float x) {
  return 0.5f * x * (1.0f + erff(x * 0.7071067811865475f));
}

__device__ __forceinline__ v16h combine8(v8h lo, v8h hi) {
  v16h r;
#pragma unroll
  for (int i = 0; i < 8; ++i) { r[i] = lo[i]; r[8 + i] = hi[i]; }
  return r;
}

__device__ __forceinline__ v4h cvt4(v4f t) {
  v4h r;
#pragma unroll
  for (int i = 0; i < 4; ++i) r[i] = (_Float16)t[i];
  return r;
}

// ---------------------------------------------------------------------------
// Generic tiled WMMA GEMM: C[M x Nc] = act(A[M x K] * B[K x Nc] + bias)
// A row-major (lda=K), B row-major (ldb=Nc).  M%64==0, Nc%64==0, K%32==0.
// Block: 256 threads (8 waves). Tile: BM=64, BN=64, BK=32.
// As is row-major [m][k] (A-fragment: two 16B runs per lane).
// Bs is stored TRANSPOSED [n][k] (B-fragment: one 32B run per lane).
// Strides padded to 40 halves (80 B) to keep 16B alignment + spread banks.
// ---------------------------------------------------------------------------
#define BM 64
#define BN 64
#define BK 32
#define KPAD (BK + 8)

__global__ __launch_bounds__(256)
void gemm_bias_kernel(const float* __restrict__ A, const float* __restrict__ B,
                      const float* __restrict__ bias, float* __restrict__ C,
                      int M, int Nc, int K, int act)
{
  __shared__ alignas(16) _Float16 As [BM][KPAD];
  __shared__ alignas(16) _Float16 BsT[BN][KPAD];

  const int tid   = threadIdx.x;
  const int wave  = tid >> 5;
  const int lane  = tid & 31;
  const int lhalf = lane >> 4;     // 0: lanes 0-15, 1: lanes 16-31
  const int l16   = lane & 15;

  const int bm = blockIdx.x * BM;
  const int bn = blockIdx.y * BN;
  const int wm = (wave & 3) * 16;  // wave row slab inside tile
  const int wn = (wave >> 2) * 32; // wave col slab inside tile

  v8f acc0 = {};
  v8f acc1 = {};

  for (int k0 = 0; k0 < K; k0 += BK) {
    // Stage A tile (64x32) as float4 loads -> packed v4h LDS stores.
#pragma unroll
    for (int j = 0; j < 2; ++j) {
      int idx = tid + j * 256;               // 0..511 float4s
      int r = idx >> 3, c4 = (idx & 7) << 2; // row, col(4-aligned)
      v4f t = *(const v4f*)&A[(size_t)(bm + r) * K + k0 + c4];
      *(v4h*)&As[r][c4] = cvt4(t);
    }
    // Stage B tile (32x64) transposed: coalesced float4 global reads,
    // scalar transposed LDS writes.
#pragma unroll
    for (int j = 0; j < 2; ++j) {
      int idx = tid + j * 256;
      int r = idx >> 4, c4 = (idx & 15) << 2; // k-row, n-col(4-aligned)
      v4f t = *(const v4f*)&B[(size_t)(k0 + r) * Nc + bn + c4];
#pragma unroll
      for (int i = 0; i < 4; ++i) BsT[c4 + i][r] = (_Float16)t[i];
    }
    __syncthreads();

    // A fragment: lane = row M=l16(+wm); halves = K run
    // [lhalf*16, +16)  -> two ds_load_b128.
    const _Float16* apt = &As[wm + l16][lhalf << 4];
    v16h af = combine8(*(const v8h*)apt, *(const v8h*)(apt + 8));

    // B fragments: lane = col N=l16(+wn[,+16]); halves = K run
    // [lhalf*16, +16)  -> two ds_load_b128 (contiguous 32B).
    const _Float16* bp0 = &BsT[wn + l16][lhalf << 4];
    const _Float16* bp1 = &BsT[wn + 16 + l16][lhalf << 4];
    v16h bf0 = combine8(*(const v8h*)bp0, *(const v8h*)(bp0 + 8));
    v16h bf1 = combine8(*(const v8h*)bp1, *(const v8h*)(bp1 + 8));

    acc0 = WMMA16(af, bf0, acc0);
    acc1 = WMMA16(af, bf1, acc1);
    __syncthreads();
  }

  // Epilogue. C layout: lane group g=lhalf holds rows v+8g, col = base+l16.
#pragma unroll
  for (int v = 0; v < 8; ++v) {
    int row = bm + wm + v + (lhalf << 3);
    int c0  = bn + wn + l16;
    int c1  = c0 + 16;
    float x0 = acc0[v] + bias[c0];
    float x1 = acc1[v] + bias[c1];
    if (act == 1) { x0 = gelu_exact(x0); x1 = gelu_exact(x1); }
    C[(size_t)row * Nc + c0] = x0;
    C[(size_t)row * Nc + c1] = x1;
  }
}

// ---------------------------------------------------------------------------
// Edge stage: degree count + scatter-add of hs[src] into aggraw[dst].
// ---------------------------------------------------------------------------
__global__ __launch_bounds__(256)
void deg_kernel(const int* __restrict__ dst, float* __restrict__ deg, int E)
{
  int e = blockIdx.x * blockDim.x + threadIdx.x;
  if (e < E) atomicAdd(&deg[dst[e]], 1.0f);
}

__global__ __launch_bounds__(256)
void scatter_kernel(const int* __restrict__ src, const int* __restrict__ dst,
                    const float* __restrict__ hs, float* __restrict__ aggraw, int E)
{
  // One wave per edge; each lane handles 8 contiguous channels (32*8 = 256).
  long long t = (long long)blockIdx.x * blockDim.x + threadIdx.x;
  if (t >= (long long)E * 32) return;
  int e  = (int)(t >> 5);
  int c0 = (int)(t & 31) << 3;
  int s = src[e], d = dst[e];
  const float* hp = hs + (size_t)s * H_DIM + c0;
  float* ap = aggraw + (size_t)d * H_DIM + c0;
  v4f x0 = *(const v4f*)hp;
  v4f x1 = *(const v4f*)(hp + 4);
#pragma unroll
  for (int j = 0; j < 4; ++j) atomicAdd(&ap[j], x0[j]);
#pragma unroll
  for (int j = 0; j < 4; ++j) atomicAdd(&ap[4 + j], x1[j]);
}

__global__ __launch_bounds__(256)
void finalize_agg_kernel(const float* __restrict__ aggraw, const float* __restrict__ ht,
                         const float* __restrict__ deg, float* __restrict__ agg, int total)
{
  int t = blockIdx.x * blockDim.x + threadIdx.x;
  if (t >= total) return;
  int n = t >> 8;                       // / H_DIM
  float dg = deg[n];
  agg[t] = (aggraw[t] + dg * ht[t]) / fmaxf(dg, 1.0f);
}

// ---------------------------------------------------------------------------
// Flash attention: batch=1, seq=N=4096, 4 heads, DH=64.
// Block = 256 threads (8 waves); each wave owns one 16-row query tile;
// block covers 128 query rows of one head.
//   Ks  row-major [key][d]   (score B-fragment: contiguous in d)
//   VsT transposed [d][key]  (PV    B-fragment: contiguous in key)
//   Ps  row-major [q][key]   (P A-fragment: two contiguous 16B runs)
// ---------------------------------------------------------------------------
__global__ __launch_bounds__(256)
void attn_kernel(const float* __restrict__ q, const float* __restrict__ k,
                 const float* __restrict__ v, float* __restrict__ ctx, int N)
{
  __shared__ alignas(16) _Float16 Ks [32][DH + 8];   // stride 72h = 144B
  __shared__ alignas(16) _Float16 VsT[DH][32 + 8];   // stride 40h = 80B
  __shared__ alignas(16) _Float16 Ps [8][16][32 + 8];

  const int tid   = threadIdx.x;
  const int wave  = tid >> 5;
  const int lane  = tid & 31;
  const int lhalf = lane >> 4;
  const int l16   = lane & 15;

  const int head  = blockIdx.y;
  const int hoff  = head * DH;
  const int qbase = blockIdx.x * 128 + wave * 16;

  // Preload the wave's Q A-fragments (16 rows x DH=64 -> two K=32 fragments),
  // pre-scaled by 1/sqrt(DH).  Four aligned float4 loads per fragment.
  const float scale = 0.125f;
  v16h qa0, qa1;
  {
    const float* qr = q + (size_t)(qbase + l16) * H_DIM + hoff;
    const int o = lhalf << 3;
    v4f a0 = *(const v4f*)(qr + o);
    v4f a1 = *(const v4f*)(qr + o + 4);
    v4f a2 = *(const v4f*)(qr + 16 + o);
    v4f a3 = *(const v4f*)(qr + 16 + o + 4);
    v4f b0 = *(const v4f*)(qr + 32 + o);
    v4f b1 = *(const v4f*)(qr + 32 + o + 4);
    v4f b2 = *(const v4f*)(qr + 48 + o);
    v4f b3 = *(const v4f*)(qr + 48 + o + 4);
#pragma unroll
    for (int i = 0; i < 4; ++i) {
      qa0[i]      = (_Float16)(a0[i] * scale);
      qa0[4 + i]  = (_Float16)(a1[i] * scale);
      qa0[8 + i]  = (_Float16)(a2[i] * scale);
      qa0[12 + i] = (_Float16)(a3[i] * scale);
      qa1[i]      = (_Float16)(b0[i] * scale);
      qa1[4 + i]  = (_Float16)(b1[i] * scale);
      qa1[8 + i]  = (_Float16)(b2[i] * scale);
      qa1[12 + i] = (_Float16)(b3[i] * scale);
    }
  }

  float mrow[8], lrow[8];
#pragma unroll
  for (int r = 0; r < 8; ++r) { mrow[r] = -1e30f; lrow[r] = 0.0f; }
  v8f acc[4] = {};

  for (int kb = 0; kb < N; kb += 32) {
    __syncthreads();  // previous iteration's LDS reads are done
    // Stage K (row-major) and V (transposed): float4 loads, 2 per thread each.
#pragma unroll
    for (int j = 0; j < 2; ++j) {
      int idx = tid + j * 256;                 // 0..511 float4s
      int r = idx >> 4, c4 = (idx & 15) << 2;  // key, d(4-aligned)
      v4f tk = *(const v4f*)&k[(size_t)(kb + r) * H_DIM + hoff + c4];
      v4f tv = *(const v4f*)&v[(size_t)(kb + r) * H_DIM + hoff + c4];
      *(v4h*)&Ks[r][c4] = cvt4(tk);
#pragma unroll
      for (int i = 0; i < 4; ++i) VsT[c4 + i][r] = (_Float16)tv[i];
    }
    __syncthreads();

    // S = Q * K^T : two 16x16 score tiles (keys +0..15, +16..31),
    // each needs 2 WMMAs over the DH=64 contraction.
    v8f s0 = {}, s1 = {};
    {
      const _Float16* kp0 = &Ks[l16][lhalf << 3];       // keys 0..15
      const _Float16* kp1 = &Ks[16 + l16][lhalf << 3];  // keys 16..31
      v16h b0 = combine8(*(const v8h*)kp0, *(const v8h*)(kp0 + 16));
      v16h b1 = combine8(*(const v8h*)kp1, *(const v8h*)(kp1 + 16));
      s0 = WMMA16(qa0, b0, s0);
      s1 = WMMA16(qa0, b1, s1);
      b0 = combine8(*(const v8h*)(kp0 + 32), *(const v8h*)(kp0 + 48));
      b1 = combine8(*(const v8h*)(kp1 + 32), *(const v8h*)(kp1 + 48));
      s0 = WMMA16(qa1, b0, s0);
      s1 = WMMA16(qa1, b1, s1);
    }

    // Online softmax. Row r of the C tile lives in VGPR r across the 16
    // lanes of one half (actual row = r + 8*lhalf): per-row reductions are
    // xor-shuffles with masks 8,4,2,1 (stay inside the 16-lane half).
    float alpha[8];
#pragma unroll
    for (int r = 0; r < 8; ++r) {
      float mx = fmaxf(s0[r], s1[r]);
#pragma unroll
      for (int m = 8; m >= 1; m >>= 1) mx = fmaxf(mx, __shfl_xor(mx, m, 32));
      float mnew = fmaxf(mrow[r], mx);
      alpha[r] = __expf(mrow[r] - mnew);
      float p0 = __expf(s0[r] - mnew);
      float p1 = __expf(s1[r] - mnew);
      s0[r] = p0; s1[r] = p1;
      float sum = p0 + p1;
#pragma unroll
      for (int m = 8; m >= 1; m >>= 1) sum += __shfl_xor(sum, m, 32);
      lrow[r] = lrow[r] * alpha[r] + sum;
      mrow[r] = mnew;
    }

    // Rescale accumulators (same row<->lane alignment as the score tile).
#pragma unroll
    for (int nb = 0; nb < 4; ++nb)
#pragma unroll
      for (int r = 0; r < 8; ++r) acc[nb][r] *= alpha[r];

    // Stage P (16x32) to LDS row-major, reload in A layout (two 16B runs).
#pragma unroll
    for (int r = 0; r < 8; ++r) {
      Ps[wave][r + (lhalf << 3)][l16]      = (_Float16)s0[r];
      Ps[wave][r + (lhalf << 3)][16 + l16] = (_Float16)s1[r];
    }
    __syncthreads();

    const _Float16* pp = &Ps[wave][l16][lhalf << 3];
    v16h pa = combine8(*(const v8h*)pp, *(const v8h*)(pp + 16));

    // ctx += P * V : 4 output column tiles of 16 over DH=64.
#pragma unroll
    for (int nb = 0; nb < 4; ++nb) {
      const _Float16* vp = &VsT[nb * 16 + l16][lhalf << 3];
      v16h bvf = combine8(*(const v8h*)vp, *(const v8h*)(vp + 16));
      acc[nb] = WMMA16(pa, bvf, acc[nb]);
    }
  }

  // Normalize by row sums and write ctx (f32, row-major N x H).
#pragma unroll
  for (int nb = 0; nb < 4; ++nb)
#pragma unroll
    for (int r = 0; r < 8; ++r) {
      int row = qbase + r + (lhalf << 3);
      int col = hoff + nb * 16 + l16;
      ctx[(size_t)row * H_DIM + col] = acc[nb][r] / lrow[r];
    }
}

// ---------------------------------------------------------------------------
// Fused residual + LayerNorm: out = LN(a + b) * g + be.  One wave per row,
// each lane owns 8 contiguous channels; reductions via xor-shuffle over 32.
// ---------------------------------------------------------------------------
__global__ __launch_bounds__(256)
void add_ln_kernel(const float* __restrict__ a, const float* __restrict__ b,
                   const float* __restrict__ g, const float* __restrict__ be,
                   float* __restrict__ out)
{
  const int wave = threadIdx.x >> 5;
  const int lane = threadIdx.x & 31;
  const int row  = blockIdx.x * 8 + wave;
  const size_t base = (size_t)row * H_DIM + lane * 8;

  v4f a0 = *(const v4f*)(a + base);
  v4f a1 = *(const v4f*)(a + base + 4);
  v4f b0 = *(const v4f*)(b + base);
  v4f b1 = *(const v4f*)(b + base + 4);

  float x[8];
  float sum = 0.0f;
#pragma unroll
  for (int j = 0; j < 4; ++j) { x[j] = a0[j] + b0[j]; x[4 + j] = a1[j] + b1[j]; }
#pragma unroll
  for (int j = 0; j < 8; ++j) sum += x[j];
#pragma unroll
  for (int m = 16; m >= 1; m >>= 1) sum += __shfl_xor(sum, m, 32);
  float mu = sum * (1.0f / H_DIM);

  float var = 0.0f;
#pragma unroll
  for (int j = 0; j < 8; ++j) { float d = x[j] - mu; var += d * d; }
#pragma unroll
  for (int m = 16; m >= 1; m >>= 1) var += __shfl_xor(var, m, 32);
  float rinv = rsqrtf(var * (1.0f / H_DIM) + 1e-5f);

  v4f g0 = *(const v4f*)(g + lane * 8);
  v4f g1 = *(const v4f*)(g + lane * 8 + 4);
  v4f e0 = *(const v4f*)(be + lane * 8);
  v4f e1 = *(const v4f*)(be + lane * 8 + 4);
  v4f o0, o1;
#pragma unroll
  for (int j = 0; j < 4; ++j) {
    o0[j] = (x[j] - mu) * rinv * g0[j] + e0[j];
    o1[j] = (x[4 + j] - mu) * rinv * g1[j] + e1[j];
  }
  *(v4f*)(out + base)     = o0;
  *(v4f*)(out + base + 4) = o1;
}

// ---------------------------------------------------------------------------
// Host-side orchestration.
// ---------------------------------------------------------------------------
extern "C" void kernel_launch(void* const* d_in, const int* in_sizes, int n_in,
                              void* d_out, int out_size, void* d_ws, size_t ws_size,
                              hipStream_t stream)
{
  (void)in_sizes; (void)n_in; (void)out_size; (void)ws_size;

  const float* h     = (const float*)d_in[0];
  const int*   eidx  = (const int*)d_in[1];     // (2, E) int32
  const float* W_src = (const float*)d_in[2];
  const float* b_src = (const float*)d_in[3];
  const float* W_tgt = (const float*)d_in[4];
  const float* b_tgt = (const float*)d_in[5];
  const float* Wq    = (const float*)d_in[6];
  const float* bq    = (const float*)d_in[7];
  const float* Wk    = (const float*)d_in[8];
  const float* bk    = (const float*)d_in[9];
  const float* Wv    = (const float*)d_in[10];
  const float* bv    = (const float*)d_in[11];
  const float* Wo    = (const float*)d_in[12];
  const float* bo    = (const float*)d_in[13];
  const float* W1    = (const float*)d_in[14];
  const float* b1    = (const float*)d_in[15];
  const float* W2    = (const float*)d_in[16];
  const float* b2    = (const float*)d_in[17];
  const float* g1    = (const float*)d_in[18];
  const float* be1   = (const float*)d_in[19];
  const float* g2    = (const float*)d_in[20];
  const float* be2   = (const float*)d_in[21];

  const int* src = eidx;
  const int* dst = eidx + E_EDGES;

  const size_t NH = (size_t)N_NODES * H_DIM;   // 1,048,576 floats
  float* ws      = (float*)d_ws;
  float* hs      = ws + 0 * NH;
  float* ht      = ws + 1 * NH;
  float* qb      = ws + 2 * NH;
  float* aggraw  = ws + 3 * NH;
  float* agg     = ws + 4 * NH;
  float* kb      = ws + 5 * NH;
  float* vb      = ws + 6 * NH;
  float* ctx     = ws + 7 * NH;
  float* attnout = ws + 8 * NH;
  float* xb      = ws + 9 * NH;
  float* f1      = ws + 10 * NH;               // N x 512 (2*NH)
  float* f2      = ws + 12 * NH;
  float* deg     = ws + 13 * NH;               // N floats

  hipMemsetAsync(aggraw, 0, NH * sizeof(float), stream);
  hipMemsetAsync(deg, 0, N_NODES * sizeof(float), stream);

  dim3 blk(256);
  dim3 gemmHH(N_NODES / BM, H_DIM / BN);        // (64, 4)
  dim3 gemmH2H(N_NODES / BM, (2 * H_DIM) / BN); // (64, 8)

  // Node-level edge linears + queries (algebraic hoist of per-edge GEMMs).
  gemm_bias_kernel<<<gemmHH, blk, 0, stream>>>(h, W_src, b_src, hs, N_NODES, H_DIM, H_DIM, 0);
  gemm_bias_kernel<<<gemmHH, blk, 0, stream>>>(h, W_tgt, b_tgt, ht, N_NODES, H_DIM, H_DIM, 0);
  gemm_bias_kernel<<<gemmHH, blk, 0, stream>>>(h, Wq, bq, qb, N_NODES, H_DIM, H_DIM, 0);

  // Edge scatter: agg[n] = (sum hs[src] + deg[n]*ht[n]) / max(deg,1)
  deg_kernel<<<(E_EDGES + 255) / 256, blk, 0, stream>>>(dst, deg, E_EDGES);
  {
    long long threads = (long long)E_EDGES * 32;
    scatter_kernel<<<(unsigned)((threads + 255) / 256), blk, 0, stream>>>(src, dst, hs, aggraw, E_EDGES);
  }
  finalize_agg_kernel<<<(unsigned)((NH + 255) / 256), blk, 0, stream>>>(aggraw, ht, deg, agg, (int)NH);

  // K / V projections from aggregated features.
  gemm_bias_kernel<<<gemmHH, blk, 0, stream>>>(agg, Wk, bk, kb, N_NODES, H_DIM, H_DIM, 0);
  gemm_bias_kernel<<<gemmHH, blk, 0, stream>>>(agg, Wv, bv, vb, N_NODES, H_DIM, H_DIM, 0);

  // Multi-head flash attention.
  attn_kernel<<<dim3(N_NODES / 128, N_HEADS), blk, 0, stream>>>(qb, kb, vb, ctx, N_NODES);

  // Output projection, residual + LN1.
  gemm_bias_kernel<<<gemmHH, blk, 0, stream>>>(ctx, Wo, bo, attnout, N_NODES, H_DIM, H_DIM, 0);
  add_ln_kernel<<<N_NODES / 8, blk, 0, stream>>>(h, attnout, g1, be1, xb);

  // FFN (exact GELU) + residual + LN2 -> output.
  gemm_bias_kernel<<<gemmH2H, blk, 0, stream>>>(xb, W1, b1, f1, N_NODES, 2 * H_DIM, H_DIM, 1);
  gemm_bias_kernel<<<gemmHH, blk, 0, stream>>>(f1, W2, b2, f2, N_NODES, H_DIM, 2 * H_DIM, 0);
  add_ln_kernel<<<N_NODES / 8, blk, 0, stream>>>(xb, f2, g2, be2, (float*)d_out);
}
